// SimilarityWeightGenerator_57415122812955
// MI455X (gfx1250) — matrix operand order repaired and checked by reading, not compiled, not run on hardware
//
#include <hip/hip_runtime.h>
#include <cstdint>
#include <cstddef>

// Problem constants (from reference): B=8, C=16, N=65536, K_CAND=10, K_TOP=4
#define B_   8
#define C_   16
#define N_   65536
#define KC   10
#define KT   4
#define NT   128                 // n positions per block (8 waves * 16)
#define NW   16                  // n positions per wave (one WMMA tile of rows)
#define SC   (NT * KC)           // floats per c-slab in LDS = 1280

typedef __attribute__((ext_vector_type(16))) _Float16 v16h;
typedef __attribute__((ext_vector_type(8)))  float    v8f;

// Low 32 bits of a flat LDS pointer are the LDS byte offset (ISA 10.2 aperture rule).
__device__ __forceinline__ uint32_t lds_off_u32(const void* p) {
  return (uint32_t)(uintptr_t)p;
}

__global__ __launch_bounds__(256)
void simweight_kernel(const float* __restrict__ orig,
                      const float* __restrict__ smp,
                      const float* __restrict__ w1,
                      const float* __restrict__ b1p,
                      const float* __restrict__ w2,
                      const float* __restrict__ b2p,
                      float* __restrict__ out)
{
  __shared__ float sS[C_ * SC];        // 81920 B: sampled tile [c][n_local][k]
  __shared__ float sLG[8][2][NW];      // per-wave logit bounce: [wave][col][m]

  const int tid   = threadIdx.x;
  const int lane  = tid & 31;
  const int wave  = tid >> 5;
  const int b     = blockIdx.x >> 9;   // 512 tiles per batch (65536/128)
  const int tile  = blockIdx.x & 511;
  const int nbase = tile * NT;

  // ---------------- async stage: sampled[b, :, nbase..+127, :] -> LDS -------
  // 16 threads per c-slab; each thread issues 20 x 16B async copies.
  {
    const int c16 = tid >> 4;          // c = 0..15
    const int l16 = tid & 15;
    const uint8_t* gsrc =
        (const uint8_t*)(smp + ((size_t)(b * C_ + c16) * N_ + nbase) * KC);
    const uint32_t ldst = lds_off_u32(&sS[c16 * SC]);
#pragma unroll
    for (int j = 0; j < 20; ++j) {
      const uint32_t byt = (uint32_t)(j * 16 + l16) * 16u;
      const uint64_t ga  = (uint64_t)(uintptr_t)(gsrc + byt);
      const uint32_t la  = ldst + byt;
      asm volatile("global_load_async_to_lds_b128 %0, %1, off"
                   :: "v"(la), "v"(ga) : "memory");
    }
  }

  // ---------------- independent setup overlapped with the async DMA --------
  const int col  = lane & 15;          // WMMA D column owned by this lane
  const int hi   = lane >> 4;          // lane half (selects K/M sub-range)
  const int coff = hi * 8;             // c-range handled by this lane
  const int row  = col;                // WMMA A row == n index within wave tile
  const int ng   = nbase + wave * NW + row;   // global n for this lane

  // original features for (c = coff..coff+7, n = ng); streamed once -> NT loads
  float og[8];
#pragma unroll
  for (int j = 0; j < 8; ++j)
    og[j] = __builtin_nontemporal_load(&orig[(size_t)(b * C_ + coff + j) * N_ + ng]);

  // B fragment (32x16 f16): column 0 = w1[0..31], column 1 = w2[0..31], rest 0.
// Lane l holds column (l&15); element i holds K = 8*(l>>4) + i for i<8, 16 + 8*(l>>4) + (i-8) for i>=8.
  v16h bf = {};
  if (col < 2) {
    const float* w = col ? w2 : w1;
#pragma unroll
    for (int i = 0; i < 16; ++i) bf[i] = (_Float16)w[(i < 8) ? (hi * 8 + i) : (8 + hi * 8 + i)];
  }

  // C accumulator (16x16 f32): bias b1 in column 0, b2 in column 1.
  const float bias = (col == 0) ? b1p[0] : ((col == 1) ? b2p[0] : 0.f);
  v8f cf;
#pragma unroll
  for (int r = 0; r < 8; ++r) cf[r] = bias;

  // orig half of the A fragment in f16, reused for all 10 candidates
  _Float16 ogh[8];
#pragma unroll
  for (int j = 0; j < 8; ++j) ogh[j] = (_Float16)og[j];

  // ---------------- complete the DMA, make tile visible to all waves -------
  asm volatile("s_wait_asynccnt 0" ::: "memory");
  __syncthreads();

  // ---------------- one WMMA per candidate: D[n,0]=logit1, D[n,1]=logit2 ---
  float L1[KC], L2[KC];
#pragma unroll
  for (int k = 0; k < KC; ++k) {
    v16h af;
#pragma unroll
    for (int j = 0; j < 8; ++j) af[j] = ogh[j];                 // K part: orig
#pragma unroll
    for (int j = 0; j < 8; ++j)                                  // K part: sampled
      af[8 + j] = (_Float16)sS[(coff + j) * SC + row * KC + k];

    v8f d = __builtin_amdgcn_wmma_f32_16x16x32_f16(
        /*neg_a=*/false, af, /*neg_b=*/false, bf,
        /*c_mod=*/(short)0, cf, /*reuse_a=*/false, /*reuse_b=*/false);

    // Only columns 0/1 carry data: lanes 0,16 -> logit1 rows, lanes 1,17 -> logit2.
    if (col < 2) {
#pragma unroll
      for (int r = 0; r < 8; ++r) sLG[wave][col][hi * 8 + r] = d[r];
    }
    // Cross-lane handoff through LDS: hardware keeps same-wave DS ops in
    // order, but the compiler must NOT hoist the broadcast loads above the
    // exec-masked stores (it did exactly that without this fence).
    __builtin_amdgcn_wave_barrier();
    L1[k] = sLG[wave][0][row];
    L2[k] = sLG[wave][1][row];
    // ...and keep the next candidate's stores below these loads.
    __builtin_amdgcn_wave_barrier();
  }

  // ---------------- branch-free top-4 (sigmoid is monotonic -> rank logits) -
  float wk[KC];
#pragma unroll
  for (int k = 0; k < KC; ++k) {
    int rank = 0;
#pragma unroll
    for (int j = 0; j < KC; ++j)
      rank += ((L1[j] > L1[k]) || ((L1[j] == L1[k]) && (j < k))) ? 1 : 0;
    const float recal = 1.f / (1.f + __expf(-L2[k]));
    wk[k] = (rank < KT) ? recal : 0.f;
  }

  // ---------------- out[c,n] = orig + sum_k sampled[c,n,k] * wk[k] ---------
#pragma unroll
  for (int j = 0; j < 8; ++j) {
    const float* sp = &sS[(coff + j) * SC + row * KC];
    float acc = og[j];
#pragma unroll
    for (int k = 0; k < KC; ++k) acc += sp[k] * wk[k];
    __builtin_nontemporal_store(acc, &out[(size_t)(b * C_ + coff + j) * N_ + ng]);
  }
}

extern "C" void kernel_launch(void* const* d_in, const int* in_sizes, int n_in,
                              void* d_out, int out_size, void* d_ws, size_t ws_size,
                              hipStream_t stream) {
  (void)in_sizes; (void)n_in; (void)out_size; (void)d_ws; (void)ws_size;
  const float* orig = (const float*)d_in[0];
  const float* smp  = (const float*)d_in[1];
  const float* w1   = (const float*)d_in[2];
  const float* b1   = (const float*)d_in[3];
  const float* w2   = (const float*)d_in[4];
  const float* b2   = (const float*)d_in[5];
  dim3 grid(B_ * (N_ / NT));   // 4096 blocks of 256 threads (8 waves)
  simweight_kernel<<<grid, dim3(256), 0, stream>>>(orig, smp, w1, b1, w2, b2,
                                                   (float*)d_out);
}
